// MetaBaseline_47236050321510
// MI455X (gfx1250) — matrix operand order repaired and checked by reading, not compiled, not run on hardware
//
#include <hip/hip_runtime.h>

typedef __attribute__((ext_vector_type(16))) _Float16 v16h;
typedef __attribute__((ext_vector_type(8)))  _Float16 v8h;
typedef __attribute__((ext_vector_type(8)))  float    v8f;

#define B_    32
#define QN_   75
#define QP_   80   // padded query rows (multiple of 16)
#define WAY_  5
#define WP_   16   // padded way rows
#define SHOT_ 5
#define HW_   25
#define C_    640

__device__ __forceinline__ float wave_reduce_add(float v) {
#pragma unroll
  for (int m = 16; m >= 1; m >>= 1) v += __shfl_xor(v, m, 32);
  return v;
}

// ---------------------------------------------------------------------------
// Kernel A: query_mean[b,q,c] = mean_hw( in1 / ||in1||_c ), stored as f16 into
// padded qmh[b][80][640]. One block per (b,q), 8 waves, b128 loads, shfl-xor
// norm reduction, deterministic LDS combine.
// ---------------------------------------------------------------------------
__global__ __launch_bounds__(256) void qmean_kernel(const float* __restrict__ in1,
                                                    _Float16* __restrict__ qmh) {
  __shared__ __align__(16) float waveacc[8][C_];
  const int bq = blockIdx.x;
  const int b = bq / QN_;
  const int q = bq - b * QN_;
  const size_t base = (size_t)bq * (size_t)(HW_ * C_);
  const int wid = threadIdx.x >> 5;
  const int lane = threadIdx.x & 31;

  float accv[20];
#pragma unroll
  for (int j = 0; j < 20; ++j) accv[j] = 0.f;

  for (int p = wid; p < HW_; p += 8) {
    const float4* dp = (const float4*)(in1 + base + (size_t)p * C_);
    float4 v[5];
    float ss = 0.f;
#pragma unroll
    for (int j = 0; j < 5; ++j) {
      v[j] = dp[lane + 32 * j];
      ss += v[j].x * v[j].x + v[j].y * v[j].y + v[j].z * v[j].z + v[j].w * v[j].w;
    }
    ss = wave_reduce_add(ss);
    const float inv = rsqrtf(ss);
#pragma unroll
    for (int j = 0; j < 5; ++j) {
      accv[4 * j + 0] += v[j].x * inv;
      accv[4 * j + 1] += v[j].y * inv;
      accv[4 * j + 2] += v[j].z * inv;
      accv[4 * j + 3] += v[j].w * inv;
    }
  }
  float4* wa = (float4*)&waveacc[wid][0];
#pragma unroll
  for (int j = 0; j < 5; ++j)
    wa[lane + 32 * j] =
        make_float4(accv[4 * j + 0], accv[4 * j + 1], accv[4 * j + 2], accv[4 * j + 3]);
  __syncthreads();

  _Float16* qrow = qmh + ((size_t)b * QP_ + q) * C_;
  for (int c = threadIdx.x; c < C_; c += 256) {
    float s = 0.f;
#pragma unroll
    for (int w = 0; w < 8; ++w) s += waveacc[w][c];
    qrow[c] = (_Float16)(s * (1.f / (float)HW_));
  }
}

// Zero the 5 pad rows per batch of qmh (rows 75..79).
__global__ __launch_bounds__(256) void qpad_kernel(_Float16* __restrict__ qmh) {
  const int i = blockIdx.x * 256 + threadIdx.x;
  if (i >= B_ * (QP_ - QN_) * C_) return;
  const int b = i / ((QP_ - QN_) * C_);
  const int rem = i - b * (QP_ - QN_) * C_;
  const int r = rem / C_;
  const int c = rem - r * C_;
  qmh[((size_t)b * QP_ + QN_ + r) * C_ + c] = (_Float16)0.f;
}

// ---------------------------------------------------------------------------
// Kernel B: per (b,way,shot) block ->
//   contrib[blk,c] = (1/SHOT) * ( mean_hw(s_norm)[c] + 5*leaky(sum_p att_p*s_norm_p[c]) )
// ---------------------------------------------------------------------------
__global__ __launch_bounds__(256) void support_kernel(const float* __restrict__ in2,
                                                      const float* __restrict__ rpn_w,
                                                      const float* __restrict__ rpn_b,
                                                      float* __restrict__ contrib) {
  __shared__ __align__(16) float wrw[C_];
  __shared__ float s_inv[HW_];
  __shared__ float s_logit[HW_];
  __shared__ float s_att[HW_];
  const int blk = blockIdx.x;
  const size_t base = (size_t)blk * (size_t)(HW_ * C_);
  const int tid = threadIdx.x;
  for (int c = tid; c < C_; c += 256) wrw[c] = rpn_w[c];
  __syncthreads();

  const int wid = tid >> 5, lane = tid & 31;
  for (int p = wid; p < HW_; p += 8) {
    const float4* dp = (const float4*)(in2 + base + (size_t)p * C_);
    const float4* wp = (const float4*)wrw;
    float ss = 0.f, dw = 0.f;
#pragma unroll
    for (int j = 0; j < 5; ++j) {
      float4 v = dp[lane + 32 * j];
      float4 w = wp[lane + 32 * j];
      ss += v.x * v.x + v.y * v.y + v.z * v.z + v.w * v.w;
      dw += v.x * w.x + v.y * w.y + v.z * w.z + v.w * w.w;
    }
    ss = wave_reduce_add(ss);
    dw = wave_reduce_add(dw);
    if (lane == 0) {
      const float inv = rsqrtf(ss);
      s_inv[p] = inv;
      s_logit[p] = dw * inv + rpn_b[0];
    }
  }
  __syncthreads();

  if (tid < 32) {
    float lg = (tid < HW_) ? s_logit[tid] : -3.0e38f;
    float mx = lg;
#pragma unroll
    for (int m = 16; m >= 1; m >>= 1) mx = fmaxf(mx, __shfl_xor(mx, m, 32));
    float e = (tid < HW_) ? __expf(lg - mx) : 0.f;
    float sum = wave_reduce_add(e);
    if (tid < HW_) s_att[tid] = e / sum;
  }
  __syncthreads();

  for (int c = tid; c < C_; c += 256) {
    float sA = 0.f, sG = 0.f;
#pragma unroll
    for (int p = 0; p < HW_; ++p) {
      const float x = in2[base + (size_t)p * C_ + c];
      const float xn = x * s_inv[p];
      sA += xn;
      sG += xn * s_att[p];
    }
    const float g = (sG >= 0.f) ? sG : 0.01f * sG;
    contrib[(size_t)blk * C_ + c] =
        (1.f / (float)SHOT_) * (sA * (1.f / (float)HW_) + 5.f * g);
  }
}

// ---------------------------------------------------------------------------
// Kernel B2: feath[b][w][c] (f16, padded to 16 ways; pad ways written as 0)
// ---------------------------------------------------------------------------
__global__ __launch_bounds__(256) void feat_kernel(const float* __restrict__ contrib,
                                                   _Float16* __restrict__ feath) {
  const int i = blockIdx.x * 256 + threadIdx.x;
  if (i >= B_ * WP_ * C_) return;
  const int bw = i / C_;
  const int c = i - bw * C_;
  const int b = bw / WP_;
  const int w = bw - b * WP_;
  float s = 0.f;
  if (w < WAY_) {
#pragma unroll
    for (int sh = 0; sh < SHOT_; ++sh)
      s += contrib[(((size_t)b * WAY_ + w) * SHOT_ + sh) * C_ + c];
  }
  feath[i] = (_Float16)s;
}

// ---------------------------------------------------------------------------
// Kernel C: sim[b,q,w] = qm[b,q,:] . feat[b,w,:] via v_wmma_f32_16x16x32_f16.
// Branchless K-loop on zero-padded f16 tiles:
//   A frag: lane m = lane&15 (rows always valid in padded qmh),
//           K = half*16 + {0..15}                          -> two b128 loads.
//   B frag: lane n = lane&15 (always valid in padded feath),
//           K = half*16 + {0..15}                          -> one 32B load.
//   D: acc[r] = D[half*8 + r][n].
// ---------------------------------------------------------------------------
__global__ __launch_bounds__(32) void sim_wmma_kernel(const _Float16* __restrict__ qmh,
                                                      const _Float16* __restrict__ feath,
                                                      float* __restrict__ out) {
  const int b = blockIdx.x / 5;
  const int mt = blockIdx.x % 5;
  const int m0 = mt * 16;
  const int lane = threadIdx.x;
  const int half = lane >> 4;
  const int idx = lane & 15;

  const _Float16* arow = qmh + ((size_t)b * QP_ + m0 + idx) * C_ + half * 16;
  const _Float16* brow = feath + ((size_t)b * WP_ + idx) * C_ + half * 16;

  v8f acc = {};
#pragma unroll 4
  for (int kk = 0; kk < C_; kk += 32) {
    const v8h alo = *(const v8h*)(arow + kk);
    const v8h ahi = *(const v8h*)(arow + kk + 8);
    const v16h a = __builtin_shufflevector(alo, ahi, 0, 1, 2, 3, 4, 5, 6, 7,
                                           8, 9, 10, 11, 12, 13, 14, 15);
    const v16h bb = *(const v16h*)(brow + kk);
    acc = __builtin_amdgcn_wmma_f32_16x16x32_f16(
        /*neg_a=*/false, a, /*neg_b=*/false, bb,
        /*c_mod=*/(short)0, acc, /*reuse_a=*/false, /*reuse_b=*/false);
  }

  if (idx < WAY_) {
#pragma unroll
    for (int r = 0; r < 8; ++r) {
      const int row = m0 + half * 8 + r;
      if (row < QN_) out[((size_t)b * QN_ + row) * WAY_ + idx] = acc[r];
    }
  }
}

// ---------------------------------------------------------------------------
extern "C" void kernel_launch(void* const* d_in, const int* in_sizes, int n_in,
                              void* d_out, int out_size, void* d_ws, size_t ws_size,
                              hipStream_t stream) {
  (void)in_sizes; (void)n_in; (void)out_size; (void)ws_size;
  const float* in1   = (const float*)d_in[0];
  const float* in2   = (const float*)d_in[1];
  const float* rpn_w = (const float*)d_in[2];
  const float* rpn_b = (const float*)d_in[3];

  // Workspace layout (bytes):
  //   qmh    : 32*80*640 f16  = 3,276,800
  //   contrib: 32*5*5*640 f32 = 2,048,000
  //   feath  : 32*16*640 f16  =   655,360
  _Float16* qmh     = (_Float16*)d_ws;
  float*    contrib = (float*)((char*)d_ws + (size_t)B_ * QP_ * C_ * sizeof(_Float16));
  _Float16* feath   = (_Float16*)((char*)contrib +
                                  (size_t)B_ * WAY_ * SHOT_ * C_ * sizeof(float));
  float* out = (float*)d_out;

  qmean_kernel<<<B_ * QN_, 256, 0, stream>>>(in1, qmh);
  qpad_kernel<<<(B_ * (QP_ - QN_) * C_ + 255) / 256, 256, 0, stream>>>(qmh);
  support_kernel<<<B_ * WAY_ * SHOT_, 256, 0, stream>>>(in2, rpn_w, rpn_b, contrib);
  feat_kernel<<<(B_ * WP_ * C_ + 255) / 256, 256, 0, stream>>>(contrib, feath);
  sim_wmma_kernel<<<B_ * 5, 32, 0, stream>>>(qmh, feath, out);
}
